// CharRNN_86792699118082
// MI455X (gfx1250) — matrix operand order repaired and checked by reading, not compiled, not run on hardware
//
#include <hip/hip_runtime.h>
#include <hip/hip_bf16.h>

#define VOCAB  256
#define EMBED  30
#define HIDDEN 128
#define BATCH  512
#define SEQLEN 1024

typedef __attribute__((ext_vector_type(16))) __bf16 v16bf;
typedef __attribute__((ext_vector_type(8)))  __bf16 v8bf;
typedef __attribute__((ext_vector_type(8)))  float  v8f;

static __device__ __forceinline__ v16bf cat16(v8bf lo, v8bf hi) {
    v16bf r;
#pragma unroll
    for (int i = 0; i < 8; i++) { r[i] = lo[i]; r[8 + i] = hi[i]; }
    return r;
}

static __device__ __forceinline__ float fast_tanh(float x) {
    // tanh(x) = 1 - 2*rcp(e^{2x}+1): v_mul, v_exp_f32, v_add, v_rcp_f32, v_fma
    float e = __expf(2.0f * x);
    return 1.0f - 2.0f * __builtin_amdgcn_rcpf(e + 1.0f);
}

// ---------------------------------------------------------------------------
// P[v][n] = embedding[v] @ W_e + b_h   (folds the E=30 GEMM + bias into a LUT)
// ---------------------------------------------------------------------------
__global__ void k_pack_P(const float* __restrict__ emb,
                         const float* __restrict__ We,
                         const float* __restrict__ bh,
                         float* __restrict__ P) {
    int tid = blockIdx.x * blockDim.x + threadIdx.x;
    if (tid >= VOCAB * HIDDEN) return;
    int v = tid >> 7, n = tid & 127;
    float s = bh[n];
#pragma unroll
    for (int e = 0; e < EMBED; e++) s += emb[v * EMBED + e] * We[e * HIDDEN + n];
    P[tid] = s;
}

// ---------------------------------------------------------------------------
// bf16 weight packs:
//   whbf[n*128+k] = bf16(W_h[n][k])         (contiguous-in-k for B-frag lanes)
//   wobf[n*128+k] = bf16(W_out[k][n])       (transpose-pack so k is contiguous)
// ---------------------------------------------------------------------------
__global__ void k_pack_W(const float* __restrict__ Wh,
                         const float* __restrict__ Wout,
                         __bf16* __restrict__ whbf,
                         __bf16* __restrict__ wobf) {
    int tid = blockIdx.x * blockDim.x + threadIdx.x;
    if (tid < HIDDEN * HIDDEN) whbf[tid] = (__bf16)Wh[tid];
    int i = tid - HIDDEN * HIDDEN;
    if (i >= 0 && i < VOCAB * HIDDEN) {
        int n = i >> 7, k = i & 127;
        wobf[i] = (__bf16)Wout[k * VOCAB + n];
    }
}

// ---------------------------------------------------------------------------
// Recurrent kernel: 2 waves per 16-row batch tile, each wave owns 64 of the
// 128 hidden columns. W_h B-fragments live in REGISTERS (16 frags = 128 VGPR
// per wave) so the 1024-step serial chain never waits on W_h loads. The h
// tile round-trips through a 4 KB LDS stage, which also performs the
// D-layout -> A-layout transpose between steps. Loop body is branch-free;
// h_last is recovered from the LDS stage after the loop.
// ---------------------------------------------------------------------------
__global__ void __launch_bounds__(64)
k_rnn(const int*    __restrict__ x,
      const float*  __restrict__ hidden,
      const float*  __restrict__ P,
      const __bf16* __restrict__ whbf,
      __bf16*       __restrict__ hs,
      float*        __restrict__ hlast) {
    __shared__ __align__(16) __bf16 sH[16 * HIDDEN]; // 4 KB

    const int tid = threadIdx.x;
    const int l   = tid & 31;       // lane
    const int w   = tid >> 5;       // wave: owns n-tiles w*4 .. w*4+3
    const int bt  = blockIdx.x;     // batch tile (16 rows)
    const int ln  = l & 15;
    const int hh  = l >> 4;         // lane half
    const int aBK = hh ? 8  : 0;    // A-frag K base (ISA 16-bit A layout)
    const int bBK = hh ? 8  : 0;    // B-frag K base (same k order as A)

    // ---- load this wave's W_h B-fragments into registers (loop-invariant)
    v16bf bReg[4][4];               // [kc][nt_local]
#pragma unroll
    for (int kc = 0; kc < 4; kc++)
#pragma unroll
        for (int ntl = 0; ntl < 4; ntl++) {
            const __bf16* p =
                whbf + ((w * 4 + ntl) * 16 + ln) * HIDDEN + kc * 32 + bBK;
            bReg[kc][ntl] = cat16(*(const v8bf*)p, *(const v8bf*)(p + 16));
        }

    // ---- stage initial hidden state (f32 -> bf16): 32 elems per thread
    {
        int row = tid >> 2, col = (tid & 3) * 32;
        const float* src = hidden + (size_t)(bt * 16 + row) * HIDDEN + col;
        for (int j = 0; j < 32; j++) sH[row * HIDDEN + col + j] = (__bf16)src[j];
    }
    __syncthreads();

    for (int t = 0; t < SEQLEN; t++) {
        // A fragments of h_{t-1} (full K=128, shared across both waves)
        v16bf aF[4];
#pragma unroll
        for (int kc = 0; kc < 4; kc++) {
            int off = ln * HIDDEN + kc * 32 + aBK;
            aF[kc] = cat16(*(const v8bf*)(sH + off), *(const v8bf*)(sH + off + 16));
        }
        // token indices for this step (row m = v + 8*hh)
        int idx[8];
#pragma unroll
        for (int v = 0; v < 8; v++)
            idx[v] = x[(size_t)(bt * 16 + v + hh * 8) * SEQLEN + t];

        // C init = P[x_t] gather (embedding@W_e + b_h pre-folded)
        v8f acc[4];
#pragma unroll
        for (int ntl = 0; ntl < 4; ntl++)
#pragma unroll
            for (int v = 0; v < 8; v++)
                acc[ntl][v] = P[idx[v] * HIDDEN + (w * 4 + ntl) * 16 + ln];

        // 16 WMMA, all operands register-resident; kc-outer order keeps
        // consecutive v_wmma on independent accumulators
#pragma unroll
        for (int kc = 0; kc < 4; kc++)
#pragma unroll
            for (int ntl = 0; ntl < 4; ntl++)
                acc[ntl] = __builtin_amdgcn_wmma_f32_16x16x32_bf16(
                    false, aF[kc], false, bReg[kc][ntl], (short)0, acc[ntl],
                    false, false);

        __syncthreads(); // both waves done reading sH before overwrite

        // tanh + restage own half of h (bf16); branch-free loop body
#pragma unroll
        for (int ntl = 0; ntl < 4; ntl++)
#pragma unroll
            for (int v = 0; v < 8; v++) {
                float f = fast_tanh(acc[ntl][v]);
                int m = v + hh * 8;
                int n = (w * 4 + ntl) * 16 + ln;
                sH[m * HIDDEN + n] = (__bf16)f;
            }
        __syncthreads();

        // stream staged h tile to hs workspace (4x b128 per thread)
        {
            int row = tid >> 2, col = (tid & 3) * 32;
            __bf16* dst = hs + ((size_t)t * BATCH + bt * 16 + row) * HIDDEN + col;
#pragma unroll
            for (int j = 0; j < 32; j += 8)
                *(v8bf*)(dst + j) = *(const v8bf*)(sH + row * HIDDEN + col + j);
        }
    }

    // ---- h_last (f32) from the final staged tile, outside the hot loop
    {
        int row = tid >> 2, col = (tid & 3) * 32;
        float* dst = hlast + (size_t)(bt * 16 + row) * HIDDEN + col;
#pragma unroll
        for (int j = 0; j < 32; j++)
            dst[j] = (float)sH[row * HIDDEN + col + j];
    }
}

// ---------------------------------------------------------------------------
// Projection: logits[r,:] = hs[r,:] @ W_out + b_out, r in [0, L*B).
// 4 waves/block, each wave owns a 16x256 tile; W_out^T bf16 in 64KB LDS.
// This kernel is HBM-write bound (537 MB f32 out) — the true floor (~23us).
// ---------------------------------------------------------------------------
__global__ void __launch_bounds__(128)
k_proj(const __bf16* __restrict__ hs,
       const __bf16* __restrict__ wobf,
       const float*  __restrict__ bout,
       float*        __restrict__ out) {
    __shared__ __align__(16) __bf16 sWo[VOCAB * HIDDEN]; // 64 KB

    const int l  = threadIdx.x & 31;
    const int w  = threadIdx.x >> 5;
    const int ln = l & 15;
    const int hh = l >> 4;

    // cooperative W_out pack load: 256 elems per thread
    for (int j = 0; j < 256; j += 8)
        *(v8bf*)(sWo + threadIdx.x * 256 + j) =
            *(const v8bf*)(wobf + threadIdx.x * 256 + j);
    __syncthreads();

    const size_t r0  = ((size_t)blockIdx.x * 4 + w) * 16;
    const int    aBK = hh ? 8  : 0;
    const int    bBK = hh ? 8  : 0;

    // A fragments straight from global hs (row-major, contiguous chunks)
    v16bf aF[4];
#pragma unroll
    for (int kc = 0; kc < 4; kc++) {
        const __bf16* p = hs + (r0 + ln) * HIDDEN + kc * 32 + aBK;
        aF[kc] = cat16(*(const v8bf*)p, *(const v8bf*)(p + 16));
    }

#pragma unroll
    for (int nt = 0; nt < 16; nt++) {
        float bias = bout[nt * 16 + ln];
        v8f acc;
#pragma unroll
        for (int v = 0; v < 8; v++) acc[v] = bias;
#pragma unroll
        for (int kc = 0; kc < 4; kc++) {
            int offB = (nt * 16 + ln) * HIDDEN + kc * 32 + bBK;
            v16bf bF = cat16(*(const v8bf*)(sWo + offB),
                             *(const v8bf*)(sWo + offB + 16));
            acc = __builtin_amdgcn_wmma_f32_16x16x32_bf16(
                false, aF[kc], false, bF, (short)0, acc, false, false);
        }
#pragma unroll
        for (int v = 0; v < 8; v++)
            out[(r0 + v + hh * 8) * VOCAB + nt * 16 + ln] = acc[v];
    }
}

// ---------------------------------------------------------------------------
extern "C" void kernel_launch(void* const* d_in, const int* in_sizes, int n_in,
                              void* d_out, int out_size, void* d_ws, size_t ws_size,
                              hipStream_t stream) {
    const int*   x    = (const int*)  d_in[0];
    const float* hid  = (const float*)d_in[1];
    const float* emb  = (const float*)d_in[2];
    const float* Wh   = (const float*)d_in[3];
    const float* We   = (const float*)d_in[4];
    const float* bh   = (const float*)d_in[5];
    const float* Wout = (const float*)d_in[6];
    const float* bout = (const float*)d_in[7];
    float* out = (float*)d_out;

    // workspace layout
    char*   ws    = (char*)d_ws;
    float*  P     = (float*)(ws);                       // 128 KB
    __bf16* whbf  = (__bf16*)(ws + 131072);             // 32 KB
    __bf16* wobf  = (__bf16*)(ws + 131072 + 32768);     // 64 KB
    __bf16* hs    = (__bf16*)(ws + 262144);             // 128 MB (L*B*H bf16)
    float*  hlast = out + (size_t)SEQLEN * BATCH * VOCAB;

    k_pack_P<<<(VOCAB * HIDDEN) / 256, 256, 0, stream>>>(emb, We, bh, P);
    k_pack_W<<<(HIDDEN * HIDDEN + VOCAB * HIDDEN + 255) / 256, 256, 0, stream>>>(
        Wh, Wout, whbf, wobf);
    k_rnn<<<BATCH / 16, 64, 0, stream>>>(x, hid, P, whbf, hs, hlast);
    k_proj<<<(SEQLEN * BATCH) / 64, 128, 0, stream>>>(hs, wobf, bout, out);
}
